// ShardedAttention_47528108097808
// MI455X (gfx1250) — matrix operand order repaired and checked by reading, not compiled, not run on hardware
//
#include <hip/hip_runtime.h>
#include <hip/hip_bf16.h>
#include <math.h>

#define S_LEN    2048
#define B_SZ     2
#define H_DIM    2048
#define N_HEADS  16
#define HEAD_DIM 128

typedef __attribute__((ext_vector_type(16))) __bf16 v16bf;
typedef __attribute__((ext_vector_type(8)))  __bf16 v8bf;
typedef __attribute__((ext_vector_type(8)))  float  v8f;

__device__ __forceinline__ __bf16 f2bf(float x) { return (__bf16)x; }

// Concatenate two 8-element bf16 LDS chunks into a 16-element WMMA fragment.
__device__ __forceinline__ v16bf frag_cat(const __bf16* lo, const __bf16* hi) {
  v8bf a = *(const v8bf*)lo;
  v8bf b = *(const v8bf*)hi;
  v16bf r;
#pragma unroll
  for (int i = 0; i < 8; ++i) { r[i] = a[i]; r[i + 8] = b[i]; }
  return r;
}

// ---------------------------------------------------------------------------
// CDNA5 async memory->LDS copies (ASYNCcnt-tracked, no VGPR staging).
// INST_OFFSET is added to BOTH the LDS and global address (ISA 08 §4.4),
// so one (lds, global) pair drives a multi-b128 strided copy.
// ---------------------------------------------------------------------------
__device__ __forceinline__ void async_copy32(void* lds_dst, const void* gsrc) {
  uint32_t l = (uint32_t)(uintptr_t)lds_dst;   // low 32 bits of generic = LDS offset
  asm volatile(
      "global_load_async_to_lds_b128 %0, %1, off\n\t"
      "global_load_async_to_lds_b128 %0, %1, off offset:16"
      :: "v"(l), "v"(gsrc) : "memory");
}

__device__ __forceinline__ void async_copy128(void* lds_dst, const void* gsrc) {
  uint32_t l = (uint32_t)(uintptr_t)lds_dst;
  asm volatile(
      "global_load_async_to_lds_b128 %0, %1, off\n\t"
      "global_load_async_to_lds_b128 %0, %1, off offset:16\n\t"
      "global_load_async_to_lds_b128 %0, %1, off offset:32\n\t"
      "global_load_async_to_lds_b128 %0, %1, off offset:48\n\t"
      "global_load_async_to_lds_b128 %0, %1, off offset:64\n\t"
      "global_load_async_to_lds_b128 %0, %1, off offset:80\n\t"
      "global_load_async_to_lds_b128 %0, %1, off offset:96\n\t"
      "global_load_async_to_lds_b128 %0, %1, off offset:112"
      :: "v"(l), "v"(gsrc) : "memory");
}

__device__ __forceinline__ void wait_async_le0() {
  asm volatile("s_wait_asynccnt 0x0" ::: "memory");
}
// Async ops complete in order; <=4 outstanding means the previous tile-pair
// (4 async instructions per wave) has fully landed in LDS.
__device__ __forceinline__ void wait_async_le4() {
  asm volatile("s_wait_asynccnt 0x4" ::: "memory");
}

// ---------------------------------------------------------------------------
// fp32 -> bf16 conversion (one-time; halves all GEMM tile traffic and lets
// the whole ~117MB working set live in the 192MB global L2)
// ---------------------------------------------------------------------------
__global__ void f32_to_bf16(const float* __restrict__ in, __bf16* __restrict__ out, size_t n) {
  size_t i = (size_t)blockIdx.x * blockDim.x + threadIdx.x;
  size_t stride = (size_t)gridDim.x * blockDim.x;
  for (; i < n; i += stride) out[i] = (__bf16)in[i];
}

__global__ void copy_f32(const float* __restrict__ in, float* __restrict__ out, int n) {
  int i = blockIdx.x * blockDim.x + threadIdx.x;
  if (i < n) out[i] = in[i];
}

// ---------------------------------------------------------------------------
// Fused QKV GEMM + bias + RoPE. 128x128 tile per block, 8 waves, bf16 WMMA,
// async double-buffered LDS staging. Since 3*HD = 384 = 3*128, every 128-wide
// N tile is exactly one {q|k|v} head vector, so RoPE (hd <-> hd^64) stays
// inside the tile (epilogue bounces through LDS to cross wave tiles).
// Output: Q/K/V bf16 in [b][nh][s][hd].
// ---------------------------------------------------------------------------
__global__ __launch_bounds__(256) void gemm_qkv_rope(
    const __bf16* __restrict__ X, const __bf16* __restrict__ W,
    const float* __restrict__ bias, const float* __restrict__ freqs,
    __bf16* __restrict__ Qb, __bf16* __restrict__ Kb, __bf16* __restrict__ Vb)
{
  __shared__ alignas(16) __bf16 As[2][128][32];
  __shared__ alignas(16) __bf16 Bs[2][128][32];
  __shared__ float Cs[128][128];

  const int tid   = threadIdx.x;
  const int lane  = tid & 31, wave = tid >> 5;
  const int wm    = wave & 3, wn   = wave >> 2;      // 4x2 wave grid
  const int mlane = lane & 15, khalf = lane >> 4;
  const int m_base = blockIdx.y * 128;
  const int n_base = blockIdx.x * 128;

  v8f acc[2][4] = {};

  const int row_ld  = tid >> 1;
  const int half_ld = tid & 1;
  const __bf16* gA = X + (size_t)(m_base + row_ld) * H_DIM + half_ld * 16;
  const __bf16* gB = W + (size_t)(n_base + row_ld) * H_DIM + half_ld * 16;

  // prologue: stage k0=0 into buffer 0
  async_copy32(&As[0][row_ld][half_ld * 16], gA);
  async_copy32(&Bs[0][row_ld][half_ld * 16], gB);

  for (int k0 = 0; k0 < H_DIM; k0 += 32) {
    const int buf = (k0 >> 5) & 1;
    if (k0 + 32 < H_DIM) {
      // prefetch next tile while current computes
      async_copy32(&As[buf ^ 1][row_ld][half_ld * 16], gA + k0 + 32);
      async_copy32(&Bs[buf ^ 1][row_ld][half_ld * 16], gB + k0 + 32);
      wait_async_le4();          // current tile landed (in-order completion)
    } else {
      wait_async_le0();
    }
    __syncthreads();
#pragma unroll
    for (int mt = 0; mt < 2; ++mt) {
      const int r = wm * 32 + mt * 16 + mlane;
      v16bf afrag = frag_cat(&As[buf][r][khalf * 8], &As[buf][r][16 + khalf * 8]);
#pragma unroll
      for (int nt = 0; nt < 4; ++nt) {
        const int c = wn * 64 + nt * 16 + mlane;
        v16bf bfrag = frag_cat(&Bs[buf][c][khalf * 8], &Bs[buf][c][16 + khalf * 8]);
        acc[mt][nt] = __builtin_amdgcn_wmma_f32_16x16x32_bf16(
            false, afrag, false, bfrag, (short)0, acc[mt][nt], false, false);
      }
    }
    __syncthreads();
  }

  // fragments -> LDS (RoPE pairs columns across wave tiles)
#pragma unroll
  for (int mt = 0; mt < 2; ++mt)
#pragma unroll
    for (int nt = 0; nt < 4; ++nt)
#pragma unroll
      for (int v = 0; v < 8; ++v) {
        int r = wm * 32 + mt * 16 + ((lane < 16) ? v : v + 8);
        int c = wn * 64 + nt * 16 + mlane;
        Cs[r][c] = acc[mt][nt][v];
      }
  __syncthreads();

  const int comp = blockIdx.x % 3;      // 0=q 1=k 2=v
  const int nh   = blockIdx.x / 3;
  __bf16* dst = (comp == 0) ? Qb : (comp == 1) ? Kb : Vb;

  const int row  = tid >> 1;
  const int colh = (tid & 1) * 64;
  const int m    = m_base + row;
  const int sq   = m / B_SZ;
  const int bb   = m % B_SZ;
  const size_t head_off = (((size_t)bb * N_HEADS + nh) * S_LEN + sq) * HEAD_DIM;
#pragma unroll 4
  for (int j = 0; j < 64; ++j) {
    int hd = colh + j;
    float x = Cs[row][hd] + bias[n_base + hd];
    float y = x;
    if (comp < 2) {
      float partner = Cs[row][hd ^ 64] + bias[n_base + (hd ^ 64)];
      float f  = freqs[(size_t)sq * HEAD_DIM + hd];
      float cs = __cosf(f), sn = __sinf(f);
      float rot = (hd < 64) ? -partner : partner;
      y = x * cs + rot * sn;
    }
    dst[head_off + hd] = f2bf(y);
  }
}

// ---------------------------------------------------------------------------
// Causal flash attention. One workgroup (4 waves) owns a 64-row Q block of
// one (b,nh) head; KV iterates over 64-row blocks j0 <= q0. Q/K tiles staged
// with async-to-LDS; V is transposed through VGPRs into LDS. Online softmax
// state kept per row in LDS; each wave owns 16 Q rows.
// ---------------------------------------------------------------------------
__global__ __launch_bounds__(128) void flash_attn(
    const __bf16* __restrict__ Qb, const __bf16* __restrict__ Kb,
    const __bf16* __restrict__ Vb, __bf16* __restrict__ ctx)
{
  __shared__ alignas(16) __bf16 Qs[64][128];
  __shared__ alignas(16) __bf16 Ks[64][128];
  __shared__ alignas(16) __bf16 VTs[128][64];   // V transposed: [hd][kv]
  __shared__ float Ss[64][64];
  __shared__ float mstat[64], lstat[64], astat[64];

  const int tid   = threadIdx.x;
  const int lane  = tid & 31, wave = tid >> 5;
  const int mlane = lane & 15, khalf = lane >> 4;
  const int q0 = blockIdx.x * 64;
  const int bh = blockIdx.y;
  const int bb = bh / N_HEADS, nh = bh % N_HEADS;
  const __bf16* Qh = Qb + (size_t)bh * S_LEN * HEAD_DIM;
  const __bf16* Kh = Kb + (size_t)bh * S_LEN * HEAD_DIM;
  const __bf16* Vh = Vb + (size_t)bh * S_LEN * HEAD_DIM;
  const float scale = 0.08838834764831845f;     // 1/sqrt(128)

  const int r_ld = tid >> 1, h_ld = (tid & 1) * 64;
  async_copy128(&Qs[r_ld][h_ld], Qh + (size_t)(q0 + r_ld) * HEAD_DIM + h_ld);
  if (tid < 64) { mstat[tid] = -INFINITY; lstat[tid] = 0.f; }

  v8f o[8] = {};
  wait_async_le0();
  __syncthreads();

  for (int j0 = 0; j0 <= q0; j0 += 64) {
    async_copy128(&Ks[r_ld][h_ld], Kh + (size_t)(j0 + r_ld) * HEAD_DIM + h_ld);
    {
      __bf16 tmp[64];
      const uint4* src = (const uint4*)(Vh + (size_t)(j0 + r_ld) * HEAD_DIM + h_ld);
#pragma unroll
      for (int i = 0; i < 8; ++i) ((uint4*)tmp)[i] = src[i];
#pragma unroll
      for (int i = 0; i < 64; ++i) VTs[h_ld + i][r_ld] = tmp[i];   // transpose scatter
    }
    wait_async_le0();
    __syncthreads();

    // S = Q K^T for this wave's 16 rows x 64 kv cols
    v8f sacc[4] = {};
    const int qr = wave * 16 + mlane;
#pragma unroll
    for (int k = 0; k < 128; k += 32) {
      v16bf afrag = frag_cat(&Qs[qr][k + khalf * 8], &Qs[qr][k + 16 + khalf * 8]);
#pragma unroll
      for (int nt = 0; nt < 4; ++nt) {
        const int kv = nt * 16 + mlane;
        v16bf bfrag = frag_cat(&Ks[kv][k + khalf * 8], &Ks[kv][k + 16 + khalf * 8]);
        sacc[nt] = __builtin_amdgcn_wmma_f32_16x16x32_bf16(
            false, afrag, false, bfrag, (short)0, sacc[nt], false, false);
      }
    }
#pragma unroll
    for (int nt = 0; nt < 4; ++nt)
#pragma unroll
      for (int v = 0; v < 8; ++v)
        Ss[wave * 16 + ((lane < 16) ? v : v + 8)][nt * 16 + mlane] = sacc[nt][v];

    // online softmax: lanes 0..15 each own one of this wave's 16 rows
    if (lane < 16) {
      const int r  = wave * 16 + lane;
      const int sq = q0 + r;
      int cmax = sq - j0 + 1; if (cmax > 64) cmax = 64;   // causal
      float mprev = mstat[r];
      float mmax  = mprev;
      for (int c = 0; c < cmax; ++c) mmax = fmaxf(mmax, Ss[r][c] * scale);
      float alpha = __expf(mprev - mmax);
      float lsum  = lstat[r] * alpha;
      for (int c = 0; c < 64; ++c) {
        float p = (c < cmax) ? __expf(Ss[r][c] * scale - mmax) : 0.f;
        Ss[r][c] = p;
        lsum += p;
      }
      mstat[r] = mmax; lstat[r] = lsum; astat[r] = alpha;
    }

    // rescale running O by alpha (per-row broadcast via LDS)
    float al[8];
#pragma unroll
    for (int v = 0; v < 8; ++v) al[v] = astat[wave * 16 + ((lane < 16) ? v : v + 8)];
#pragma unroll
    for (int nt = 0; nt < 8; ++nt)
#pragma unroll
      for (int v = 0; v < 8; ++v) o[nt][v] *= al[v];

    // O += P @ V   (contraction over 64 kv = 2 WMMA k-steps)
#pragma unroll
    for (int kk = 0; kk < 2; ++kk) {
      v16bf pfrag;
#pragma unroll
      for (int i = 0; i < 16; ++i) {
        int kidx = kk * 32 + ((i >= 8) ? 16 : 0) + khalf * 8 + (i & 7);
        pfrag[i] = f2bf(Ss[qr][kidx]);
      }
#pragma unroll
      for (int nt = 0; nt < 8; ++nt) {
        const int hd = nt * 16 + mlane;
        v16bf bfrag = frag_cat(&VTs[hd][kk * 32 + khalf * 8],
                               &VTs[hd][kk * 32 + 16 + khalf * 8]);
        o[nt] = __builtin_amdgcn_wmma_f32_16x16x32_bf16(
            false, pfrag, false, bfrag, (short)0, o[nt], false, false);
      }
    }
    __syncthreads();
  }

  float linv[8];
#pragma unroll
  for (int v = 0; v < 8; ++v) linv[v] = 1.f / lstat[wave * 16 + ((lane < 16) ? v : v + 8)];
#pragma unroll
  for (int nt = 0; nt < 8; ++nt)
#pragma unroll
    for (int v = 0; v < 8; ++v) {
      int r  = wave * 16 + ((lane < 16) ? v : v + 8);
      int sq = q0 + r;
      int hd = nt * 16 + mlane;
      ctx[((size_t)(sq * B_SZ + bb)) * H_DIM + nh * HEAD_DIM + hd] = f2bf(o[nt][v] * linv[v]);
    }
}

// ---------------------------------------------------------------------------
// Output projection: ctx [4096][2048] bf16 x Wd^T -> out fp32 [s][b][h].
// skip_bias_add semantics: bias is NOT added here (returned separately).
// ---------------------------------------------------------------------------
__global__ __launch_bounds__(256) void gemm_out(
    const __bf16* __restrict__ A, const __bf16* __restrict__ W,
    float* __restrict__ out)
{
  __shared__ alignas(16) __bf16 As[2][128][32];
  __shared__ alignas(16) __bf16 Bs[2][128][32];

  const int tid   = threadIdx.x;
  const int lane  = tid & 31, wave = tid >> 5;
  const int wm    = wave & 3, wn   = wave >> 2;
  const int mlane = lane & 15, khalf = lane >> 4;
  const int m_base = blockIdx.y * 128;
  const int n_base = blockIdx.x * 128;

  v8f acc[2][4] = {};

  const int row_ld  = tid >> 1;
  const int half_ld = tid & 1;
  const __bf16* gA = A + (size_t)(m_base + row_ld) * H_DIM + half_ld * 16;
  const __bf16* gB = W + (size_t)(n_base + row_ld) * H_DIM + half_ld * 16;

  async_copy32(&As[0][row_ld][half_ld * 16], gA);
  async_copy32(&Bs[0][row_ld][half_ld * 16], gB);

  for (int k0 = 0; k0 < H_DIM; k0 += 32) {
    const int buf = (k0 >> 5) & 1;
    if (k0 + 32 < H_DIM) {
      async_copy32(&As[buf ^ 1][row_ld][half_ld * 16], gA + k0 + 32);
      async_copy32(&Bs[buf ^ 1][row_ld][half_ld * 16], gB + k0 + 32);
      wait_async_le4();
    } else {
      wait_async_le0();
    }
    __syncthreads();
#pragma unroll
    for (int mt = 0; mt < 2; ++mt) {
      const int r = wm * 32 + mt * 16 + mlane;
      v16bf afrag = frag_cat(&As[buf][r][khalf * 8], &As[buf][r][16 + khalf * 8]);
#pragma unroll
      for (int nt = 0; nt < 4; ++nt) {
        const int c = wn * 64 + nt * 16 + mlane;
        v16bf bfrag = frag_cat(&Bs[buf][c][khalf * 8], &Bs[buf][c][16 + khalf * 8]);
        acc[mt][nt] = __builtin_amdgcn_wmma_f32_16x16x32_bf16(
            false, afrag, false, bfrag, (short)0, acc[mt][nt], false, false);
      }
    }
    __syncthreads();
  }

#pragma unroll
  for (int mt = 0; mt < 2; ++mt)
#pragma unroll
    for (int nt = 0; nt < 4; ++nt)
#pragma unroll
      for (int v = 0; v < 8; ++v) {
        int r = m_base + wm * 32 + mt * 16 + ((lane < 16) ? v : v + 8);
        int c = n_base + wn * 64 + nt * 16 + mlane;
        out[(size_t)r * H_DIM + c] = acc[mt][nt][v];
      }
}

// ---------------------------------------------------------------------------
extern "C" void kernel_launch(void* const* d_in, const int* in_sizes, int n_in,
                              void* d_out, int out_size, void* d_ws, size_t ws_size,
                              hipStream_t stream) {
  (void)in_sizes; (void)n_in; (void)out_size; (void)ws_size;
  const float* hidden = (const float*)d_in[0];
  const float* freqs  = (const float*)d_in[1];
  const float* Wqkv   = (const float*)d_in[2];
  const float* bqkv   = (const float*)d_in[3];
  const float* Wd     = (const float*)d_in[4];
  const float* bd     = (const float*)d_in[5];
  float* out = (float*)d_out;

  const size_t NE_X    = (size_t)S_LEN * B_SZ * H_DIM;     // 8,388,608
  const size_t NE_WQKV = (size_t)3 * H_DIM * H_DIM;        // 12,582,912
  const size_t NE_WD   = (size_t)H_DIM * H_DIM;            // 4,194,304
  const size_t NE_HEAD = (size_t)B_SZ * N_HEADS * S_LEN * HEAD_DIM;  // 8,388,608

  __bf16* ws = (__bf16*)d_ws;
  size_t off = 0;
  __bf16* Xbf     = ws + off; off += NE_X;
  __bf16* Wqkv_bf = ws + off; off += NE_WQKV;
  __bf16* Wd_bf   = ws + off; off += NE_WD;
  __bf16* Qb      = ws + off; off += NE_HEAD;
  __bf16* Kb      = ws + off; off += NE_HEAD;
  __bf16* Vb      = ws + off; off += NE_HEAD;
  __bf16* Ctx     = ws + off; off += NE_X;

  f32_to_bf16<<<2048, 256, 0, stream>>>(hidden, Xbf, NE_X);
  f32_to_bf16<<<2048, 256, 0, stream>>>(Wqkv, Wqkv_bf, NE_WQKV);
  f32_to_bf16<<<2048, 256, 0, stream>>>(Wd, Wd_bf, NE_WD);

  gemm_qkv_rope<<<dim3(48, 32), 256, 0, stream>>>(Xbf, Wqkv_bf, bqkv, freqs, Qb, Kb, Vb);
  flash_attn<<<dim3(32, 32), 128, 0, stream>>>(Qb, Kb, Vb, Ctx);
  gemm_out<<<dim3(16, 32), 256, 0, stream>>>(Ctx, Wd_bf, out);
  copy_f32<<<8, 256, 0, stream>>>(bd, out + NE_X, H_DIM);
}
